// CausalSelfAttention_64381559767040
// MI455X (gfx1250) — matrix operand order repaired and checked by reading, not compiled, not run on hardware
//
#include <hip/hip_runtime.h>
#include <cmath>

// ---------- types ----------
typedef __bf16 bf16_t;
typedef __attribute__((ext_vector_type(16))) __bf16 v16bf;
typedef __attribute__((ext_vector_type(8)))  __bf16 v8bf;
typedef __attribute__((ext_vector_type(8)))  float  v8f;

union ABfrag { v16bf v; v8bf h[2]; };

// Scheduling fence: keep prefetch loads above, WMMAs below.
#if defined(__has_builtin)
#if __has_builtin(__builtin_amdgcn_sched_barrier)
#define SCHED_FENCE() __builtin_amdgcn_sched_barrier(0)
#endif
#endif
#ifndef SCHED_FENCE
#define SCHED_FENCE()
#endif

// problem constants
constexpr int Bsz = 2;
constexpr int Nseq = 4096;
constexpr int Dm = 768;

// ---------------------------------------------------------------------------
// fp32 -> bf16 elementwise
// ---------------------------------------------------------------------------
__global__ __launch_bounds__(256) void cvt_bf16_kernel(const float* __restrict__ src,
                                                       bf16_t* __restrict__ dst, int n) {
  int i = blockIdx.x * 256 + threadIdx.x;
  if (i < n) dst[i] = (bf16_t)src[i];
}

// fp32 [k][n] -> bf16 transposed [n][k]   (768x768)
__global__ __launch_bounds__(256) void cvt_wT_kernel(const float* __restrict__ W,
                                                     bf16_t* __restrict__ Wt) {
  int i = blockIdx.x * 256 + threadIdx.x;
  if (i < Dm * Dm) {
    int k = i / Dm, n = i % Dm;          // coalesced read over n
    Wt[n * Dm + k] = (bf16_t)W[i];
  }
}

// ---------------------------------------------------------------------------
// QKV projection:  [8192 x 768] x [768 x 768] in bf16, f32 accumulate.
// One wave per 16x64 output tile; gridDim.z selects q / k / v.
// Ping-pong double buffering with sched_barrier fences so the scheduler
// cannot sink the next-step loads below the current-step WMMAs.
// v is written TRANSPOSED as vT[b][d][token].
// ---------------------------------------------------------------------------
__global__ __launch_bounds__(256) void qkv_gemm_kernel(const bf16_t* __restrict__ xb,
                                                       const bf16_t* __restrict__ Wt,
                                                       bf16_t* __restrict__ qb,
                                                       bf16_t* __restrict__ kb,
                                                       bf16_t* __restrict__ vT) {
  const int lane = threadIdx.x & 31;
  const int wave = threadIdx.x >> 5;
  const int g    = lane >> 4;      // half-wave group
  const int nidx = lane & 15;      // column / row-in-tile index
  const int which = blockIdx.z;    // 0=q 1=k 2=v

  const int w = blockIdx.x * 8 + wave;       // 0 .. 6143
  const int mtile = w / 12;                  // 512 row tiles of 16
  const int nb    = (w % 12) * 64;           // 12 col groups of 64
  const int rowbase = mtile * 16;

  const bf16_t* Wsel  = Wt + (size_t)which * Dm * Dm;      // [n][k] layout
  const bf16_t* arow  = xb + (size_t)(rowbase + nidx) * Dm;
  const bf16_t* brow0 = Wsel + (size_t)(nb + nidx) * Dm;   // + t*16*Dm + kk + g*16

  v8f acc[4] = {};
  ABfrag a0, a1;
  v16bf  b0[4], b1[4];

  auto loadA = [&](ABfrag& a, int kk) {
    a.h[0] = *(const v8bf*)(arow + kk + g * 16);
    a.h[1] = *(const v8bf*)(arow + kk + g * 16 + 8);
  };
  auto loadB = [&](v16bf* bb, int kk) {
#pragma unroll
    for (int t = 0; t < 4; ++t)
      bb[t] = *(const v16bf*)(brow0 + (size_t)t * 16 * Dm + kk + g * 16);
  };
  auto mma4 = [&](const ABfrag& a, const v16bf* bb) {
#pragma unroll
    for (int t = 0; t < 4; ++t)
      acc[t] = __builtin_amdgcn_wmma_f32_16x16x32_bf16(
          false, a.v, false, bb[t], (short)0, acc[t], false, false);
  };

  loadA(a0, 0);
  loadB(b0, 0);
  for (int kk = 0; kk < Dm; kk += 64) {      // 12 iterations, 2 K-steps each
    loadA(a1, kk + 32);
    loadB(b1, kk + 32);
    SCHED_FENCE();                           // loads for step i+1 stay above
    mma4(a0, b0);                            // waits only on the a0/b0 group
    SCHED_FENCE();
    if (kk + 64 < Dm) {
      loadA(a0, kk + 64);
      loadB(b0, kk + 64);
    }
    SCHED_FENCE();
    mma4(a1, b1);
    SCHED_FENCE();
  }

  if (which < 2) {
    bf16_t* dst = (which == 0) ? qb : kb;
#pragma unroll
    for (int t = 0; t < 4; ++t)
#pragma unroll
      for (int r = 0; r < 8; ++r) {
        int row = rowbase + r + 8 * g;
        dst[(size_t)row * Dm + nb + t * 16 + nidx] = (bf16_t)acc[t][r];
      }
  } else {
    const int bidx = rowbase >> 12;       // batch (tiles never straddle batches)
    const int tok  = rowbase & (Nseq - 1);
#pragma unroll
    for (int t = 0; t < 4; ++t) {
      int d = nb + t * 16 + nidx;
      bf16_t* dst = vT + ((size_t)bidx * Dm + d) * Nseq + tok + 8 * g;
#pragma unroll
      for (int r = 0; r < 8; ++r) dst[r] = (bf16_t)acc[t][r];   // contiguous tokens
    }
  }
}

// ---------------------------------------------------------------------------
// Flash attention, no mask (faithful to reference).
// Block = 256 threads (8 waves) owns a 16-query tile.
// Q tile is register-resident per wave (24 A-fragments). k^T and vT
// B-fragment streams are ping-pong double buffered with sched fences.
// ---------------------------------------------------------------------------
__global__ __launch_bounds__(256) void flash_attn_kernel(const bf16_t* __restrict__ q,
                                                         const bf16_t* __restrict__ k,
                                                         const bf16_t* __restrict__ vT,
                                                         float* __restrict__ out) {
  __shared__ float  S[16][132];          // padded: rows r and r+8 hit different banks
  __shared__ bf16_t P[16][128];
  __shared__ float  Spart[16][16];
  __shared__ float  rowM[16], rowL[16], rowF[16];

  const int tid  = threadIdx.x;
  const int lane = tid & 31;
  const int wave = tid >> 5;
  const int g    = lane >> 4;
  const int nidx = lane & 15;

  const int b      = blockIdx.x >> 8;          // 256 query tiles per batch
  const int qbase  = (blockIdx.x & 255) * 16;
  const int dchunk = wave * 96;                // this wave's output d range

  if (tid < 16) { rowM[tid] = -1e30f; rowL[tid] = 0.f; }
  v8f acc[6] = {};

  const float scale = 0.03608439182435161f;    // 1/sqrt(768)

  // ---- preload the whole Q tile into registers (24 fragments) ----
  const bf16_t* qrow = q + ((size_t)(b * Nseq + qbase) + nidx) * Dm;
  ABfrag qf[24];
#pragma unroll
  for (int ds = 0; ds < 24; ++ds) {
    qf[ds].h[0] = *(const v8bf*)(qrow + ds * 32 + g * 16);
    qf[ds].h[1] = *(const v8bf*)(qrow + ds * 32 + g * 16 + 8);
  }

  // per-wave base into vT for P*V B-fragments: column d = dchunk + t*16 + nidx
  const bf16_t* vbase = vT + ((size_t)b * Dm + dchunk + nidx) * Nseq + g * 16;

  __syncthreads();

  for (int kb0 = 0; kb0 < Nseq; kb0 += 128) {
    // ---- scores: S[16q x 16key] for this wave's key subtile ----
    const bf16_t* krow =
        k + ((size_t)(b * Nseq + kb0 + wave * 16) + nidx) * Dm + g * 16;
    if (kb0 + 128 < Nseq) {  // prefetch next key tile (global_prefetch_b8)
      __builtin_prefetch(krow + 128 * Dm, 0, 1);
    }
    v8f s = {};
    {
      v16bf kf[2];                       // ping-pong, rotated by the loads
      kf[0] = *(const v16bf*)(krow);
#pragma unroll
      for (int ds = 0; ds < 24; ++ds) {
        if (ds < 23) kf[(ds + 1) & 1] = *(const v16bf*)(krow + (ds + 1) * 32);
        SCHED_FENCE();                   // next k-fragment load stays above
        s = __builtin_amdgcn_wmma_f32_16x16x32_bf16(
            false, qf[ds].v, false, kf[ds & 1], (short)0, s, false, false);
        SCHED_FENCE();
      }
    }
#pragma unroll
    for (int r = 0; r < 8; ++r)
      S[r + 8 * g][wave * 16 + nidx] = s[r] * scale;
    __syncthreads();

    // ---- parallel row-max: 256 threads reduce 8 each, 16 fold partials ----
    {
      int row = tid >> 4, c0 = (tid & 15) * 8;
      float mx8 = -1e30f;
#pragma unroll
      for (int c = 0; c < 8; ++c) mx8 = fmaxf(mx8, S[row][c0 + c]);
      Spart[row][tid & 15] = mx8;
    }
    __syncthreads();
    if (tid < 16) {
      float mx = rowM[tid];
#pragma unroll
      for (int c = 0; c < 16; ++c) mx = fmaxf(mx, Spart[tid][c]);
      rowF[tid] = __expf(rowM[tid] - mx);
      rowM[tid] = mx;
    }
    __syncthreads();

    // ---- exponentiate -> P (bf16), partial row sums ----
    {
      int row = tid >> 4, c0 = (tid & 15) * 8;
      float mrow = rowM[row], ssum = 0.f;
#pragma unroll
      for (int c = 0; c < 8; ++c) {
        float e = __expf(S[row][c0 + c] - mrow);
        P[row][c0 + c] = (bf16_t)e;
        ssum += e;
      }
      Spart[row][tid & 15] = ssum;
    }
    __syncthreads();

    if (tid < 16) {              // fold partial sums into running denominator
      float s2 = 0.f;
#pragma unroll
      for (int c = 0; c < 16; ++c) s2 += Spart[tid][c];
      rowL[tid] = rowL[tid] * rowF[tid] + s2;
    }

    // ---- rescale accumulators by rowF, then acc += P @ V ----
    float f[8];
#pragma unroll
    for (int r = 0; r < 8; ++r) f[r] = rowF[r + 8 * g];
#pragma unroll
    for (int t = 0; t < 6; ++t)
#pragma unroll
      for (int r = 0; r < 8; ++r) acc[t][r] *= f[r];

    {
      v16bf vb0[6], vb1[6];              // ping-pong across the 4 key sub-steps
      auto loadV = [&](v16bf* bb, int ks) {
#pragma unroll
        for (int t = 0; t < 6; ++t)
          bb[t] = *(const v16bf*)(vbase + (size_t)t * 16 * Nseq + kb0 + ks * 32);
      };
      auto mma6 = [&](const ABfrag& pa, const v16bf* bb) {
#pragma unroll
        for (int t = 0; t < 6; ++t)
          acc[t] = __builtin_amdgcn_wmma_f32_16x16x32_bf16(
              false, pa.v, false, bb[t], (short)0, acc[t], false, false);
      };
      auto loadP = [&](ABfrag& pa, int ks) {    // from LDS (ds_load_b128)
        pa.h[0] = *(const v8bf*)(&P[nidx][ks * 32 + g * 16]);
        pa.h[1] = *(const v8bf*)(&P[nidx][ks * 32 + g * 16 + 8]);
      };

      loadV(vb0, 0);
#pragma unroll
      for (int ks = 0; ks < 4; ks += 2) {
        ABfrag pa;
        loadV(vb1, ks + 1);
        loadP(pa, ks);
        SCHED_FENCE();                   // vb1 loads stay above vb0 WMMAs
        mma6(pa, vb0);
        SCHED_FENCE();
        if (ks + 2 < 4) loadV(vb0, ks + 2);
        loadP(pa, ks + 1);
        SCHED_FENCE();
        mma6(pa, vb1);
        SCHED_FENCE();
      }
    }
    __syncthreads();
  }

  // ---- normalize and write fp32 output ----
  float linv[8];
#pragma unroll
  for (int r = 0; r < 8; ++r) linv[r] = 1.f / rowL[r + 8 * g];
#pragma unroll
  for (int t = 0; t < 6; ++t) {
    int d = dchunk + t * 16 + nidx;
#pragma unroll
    for (int r = 0; r < 8; ++r) {
      int row = qbase + r + 8 * g;
      out[((size_t)b * Nseq + row) * Dm + d] = acc[t][r] * linv[r];
    }
  }
}

// ---------------------------------------------------------------------------
// host launcher
// ---------------------------------------------------------------------------
extern "C" void kernel_launch(void* const* d_in, const int* in_sizes, int n_in,
                              void* d_out, int out_size, void* d_ws, size_t ws_size,
                              hipStream_t stream) {
  const float* x  = (const float*)d_in[0];
  const float* Wq = (const float*)d_in[1];
  const float* Wk = (const float*)d_in[2];
  const float* Wv = (const float*)d_in[3];
  float* out = (float*)d_out;

  char* ws = (char*)d_ws;
  const size_t xElems = (size_t)Bsz * Nseq * Dm;       // 6,291,456
  const size_t wElems = (size_t)Dm * Dm;               // 589,824

  bf16_t* xb = (bf16_t*)ws;                 ws += xElems * sizeof(bf16_t);
  bf16_t* Wt = (bf16_t*)ws;                 ws += 3 * wElems * sizeof(bf16_t);
  bf16_t* qb = (bf16_t*)ws;                 ws += xElems * sizeof(bf16_t);
  bf16_t* kb = (bf16_t*)ws;                 ws += xElems * sizeof(bf16_t);
  bf16_t* vT = (bf16_t*)ws;                 ws += xElems * sizeof(bf16_t);

  // 1) convert x and transpose-convert weights to bf16
  cvt_bf16_kernel<<<(int)((xElems + 255) / 256), 256, 0, stream>>>(x, xb, (int)xElems);
  cvt_wT_kernel<<<(int)((wElems + 255) / 256), 256, 0, stream>>>(Wq, Wt);
  cvt_wT_kernel<<<(int)((wElems + 255) / 256), 256, 0, stream>>>(Wk, Wt + wElems);
  cvt_wT_kernel<<<(int)((wElems + 255) / 256), 256, 0, stream>>>(Wv, Wt + 2 * wElems);

  // 2) QKV projection (512 row-tiles * 12 col-groups / 8 waves = 768 blocks; z = q/k/v)
  dim3 gq(768, 1, 3);
  qkv_gemm_kernel<<<gq, 256, 0, stream>>>(xb, Wt, qb, kb, vT);

  // 3) flash attention: 2 batches * 256 query tiles
  flash_attn_kernel<<<512, 256, 0, stream>>>(qb, kb, vT, out);
}
